// PhysicalRecurrentLayer_27822798144119
// MI455X (gfx1250) — matrix operand order repaired and checked by reading, not compiled, not run on hardware
//
#include <hip/hip_runtime.h>
#include <hip/hip_bf16.h>

#define BB   4
#define CC   128
#define LL   24
#define HWSZ 4096
#define RR   64
#define NWAVE 4
#define NT   (NWAVE * 16)   // 64 pixels per block
#define XROW 144            // LDS row stride in shorts (288B: 32B-aligned, conflict-breaking)

typedef __attribute__((ext_vector_type(16))) __bf16 bf16x16;
typedef __attribute__((ext_vector_type(8)))  float  v8f;

__device__ __forceinline__ unsigned short f2bf(float f) {
  unsigned int u = __builtin_bit_cast(unsigned int, f);
  u += 0x7FFFu + ((u >> 16) & 1u);           // round-to-nearest-even
  return (unsigned short)(u >> 16);
}

__device__ __forceinline__ unsigned int pk2bf(float a, float b) {
#if __has_builtin(__builtin_amdgcn_cvt_pk_bf16_f32)
  auto p = __builtin_amdgcn_cvt_pk_bf16_f32(a, b);
  return __builtin_bit_cast(unsigned int, p);
#else
  return (unsigned int)f2bf(a) | ((unsigned int)f2bf(b) << 16);
#endif
}

// ---------------------------------------------------------------------------
// Prep: dr = softplus(lam); fuse Mf = U*V; pack Mf and Wout as bf16 in the
// WMMA A-fragment lane layout this code uses (k contiguous per half) so the main kernel loads fragments
// with flat 32-byte loads (the main kernel's compiler hoists all 64 fragments
// into registers for the whole recurrence).
// Fragment f = mt*4 + kt; entry (f, lane) holds 16 bf16:
//   m = mt*16 + (lane&15), half = lane>>4
//   k = kt*32 + half*16 + i
// ---------------------------------------------------------------------------
__global__ __launch_bounds__(256) void prl_prep(
    const float* __restrict__ lam, const float* __restrict__ U,
    const float* __restrict__ V,   const float* __restrict__ Wt,
    unsigned short* __restrict__ wsM, unsigned short* __restrict__ wsW,
    float* __restrict__ dr)
{
  const int tid = threadIdx.x;
  if (tid < CC) {
    float v = lam[tid];
    dr[tid] = (v > 20.f) ? v : log1pf(__expf(v));
  }
  for (int e = tid; e < 32 * 32; e += 256) {      // 32 fragments x 32 lanes
    const int f  = e >> 5, ln = e & 31;
    const int mt = f >> 2, kt = f & 3;
    const int m  = mt * 16 + (ln & 15);
    const int half = ln >> 4;
    for (int i = 0; i < 16; ++i) {
      const int k = kt * 32 + half * 16 + i;
      float s = 0.f;                               // fused M = U * V
      for (int r = 0; r < RR; ++r) s += U[m * RR + r] * V[r * CC + k];
      wsM[e * 16 + i] = f2bf(s);
      wsW[e * 16 + i] = f2bf(Wt[m * CC + k]);
    }
  }
}

// ---------------------------------------------------------------------------
// Main recurrent kernel. 128 threads = 4 waves; wave owns 16 pixel columns.
// h state: 8 v8f accumulators per lane in native WMMA D layout
//   (c = mt*16 + 8*(lane>>4) + r, n = lane&15), persistent across all t.
// Software-pipelined: step t+1's x tile is loaded (float4 each) at the top
// of step t and written to the alternate LDS buffer at the bottom.
// ---------------------------------------------------------------------------
__global__ __launch_bounds__(128) void prl_main(
    const float* __restrict__ x, const float* __restrict__ lh,
    const float* __restrict__ listT,
    const unsigned short* __restrict__ wsM, const unsigned short* __restrict__ wsW,
    const float* __restrict__ dr,
    float* __restrict__ out, float* __restrict__ hout)
{
  __shared__ __align__(32) unsigned short sXT[2][NWAVE][16][XROW]; // x tile [col][c] bf16 (dbl-buf)
  __shared__ __align__(32) unsigned short sHT[NWAVE][16][XROW];    // h tile [col][c] bf16
  __shared__ float sA[2][NWAVE][CC];                               // decay A[c] (dbl-buf)

  const int tid  = threadIdx.x;
  const int w    = tid >> 5;
  const int lane = tid & 31;
  const int col  = lane & 15;
  const int half = lane >> 4;
  const int c_off = lane >> 2;        // 0..7   (staging map)
  const int px4   = (lane & 3) * 4;   // 0,4,8,12

  const int blk  = blockIdx.x;
  const int b    = blk / (HWSZ / NT);
  const int p0   = (blk % (HWSZ / NT)) * NT + w * 16;
  const int pcol = p0 + col;

  // ---- initial hidden state (f32, register-resident, D layout) ----
  v8f h[8];
#pragma unroll
  for (int mt = 0; mt < 8; ++mt) {
    const int cb = mt * 16 + half * 8;
#pragma unroll
    for (int r = 0; r < 8; ++r)
      h[mt][r] = lh[(size_t)(b * CC + cb + r) * HWSZ + pcol];
  }

  // per-lane slice of decay rates (4 channels/lane)
  float drl[4];
#pragma unroll
  for (int j = 0; j < 4; ++j) drl[j] = dr[lane * 4 + j];

  const float* xbase = x + (size_t)b * CC * LL * HWSZ;   // + t*HWSZ + c*LL*HWSZ + p

  // ---- preload + stage tile t=0 ----
  float4 rx[16];
#pragma unroll
  for (int j = 0; j < 16; ++j) {
    const int c = j * 8 + c_off;
    rx[j] = *(const float4*)(xbase + (size_t)c * (LL * HWSZ) + p0 + px4);
  }
  {
    const float dt0 = listT[b * LL];
#pragma unroll
    for (int j = 0; j < 4; ++j) sA[0][w][lane * 4 + j] = __expf(-drl[j] * dt0);
#pragma unroll
    for (int j = 0; j < 16; ++j) {
      const int c = j * 8 + c_off;
      sXT[0][w][px4 + 0][c] = f2bf(rx[j].x);
      sXT[0][w][px4 + 1][c] = f2bf(rx[j].y);
      sXT[0][w][px4 + 2][c] = f2bf(rx[j].z);
      sXT[0][w][px4 + 3][c] = f2bf(rx[j].w);
    }
  }
  __syncthreads();

  for (int t = 0; t < LL; ++t) {
    const int cur = t & 1, nxt = cur ^ 1;

    // ---- kick off t+1's tile loads early (latency hidden by this step) ----
    if (t + 1 < LL) {
#pragma unroll
      for (int j = 0; j < 16; ++j) {
        const int c = j * 8 + c_off;
        rx[j] = *(const float4*)(xbase + (size_t)(t + 1) * HWSZ +
                                 (size_t)c * (LL * HWSZ) + p0 + px4);
      }
      if (t + 2 < LL)  // distance-2 line prefetch (global_prefetch_b8)
        __builtin_prefetch(xbase + (size_t)(t + 2) * HWSZ +
                           (size_t)(lane * 4) * (LL * HWSZ) + p0, 0, 0);
    }

    // ---- h = A*h + Mf*x_t  (decay folded into WMMA C operand) ----
#pragma unroll
    for (int mt = 0; mt < 8; ++mt) {
      const int cb = mt * 16 + half * 8;
      v8f acc;
      const float* pa = &sA[cur][w][cb];
#pragma unroll
      for (int r = 0; r < 8; ++r) acc[r] = pa[r] * h[mt][r];
#pragma unroll
      for (int kt = 0; kt < 4; ++kt) {
        bf16x16 a  = *(const bf16x16*)(const void*)(wsM + ((mt * 4 + kt) * 32 + lane) * 16);
        bf16x16 bm = *(const bf16x16*)(const void*)(&sXT[cur][w][col][kt * 32 + half * 16]);
        acc = __builtin_amdgcn_wmma_f32_16x16x32_bf16(
                  false, a, false, bm, (short)0, acc, false, false);
      }
      h[mt] = acc;
      // round h to bf16 into LDS for the Wout matmul (single 16B store)
      uint4 q;
      q.x = pk2bf(acc[0], acc[1]);
      q.y = pk2bf(acc[2], acc[3]);
      q.z = pk2bf(acc[4], acc[5]);
      q.w = pk2bf(acc[6], acc[7]);
      *(uint4*)(void*)(&sHT[w][col][cb]) = q;
    }
    __syncthreads();  // sHT cross-lane visibility

    // ---- y_t = x_t + Wout*h  (streaming stores; last-use residual loads) ----
#pragma unroll
    for (int mt = 0; mt < 8; ++mt) {
      v8f acc;
#pragma unroll
      for (int r = 0; r < 8; ++r) acc[r] = 0.f;
#pragma unroll
      for (int kt = 0; kt < 4; ++kt) {
        bf16x16 a  = *(const bf16x16*)(const void*)(wsW + ((mt * 4 + kt) * 32 + lane) * 16);
        bf16x16 bm = *(const bf16x16*)(const void*)(&sHT[w][col][kt * 32 + half * 16]);
        acc = __builtin_amdgcn_wmma_f32_16x16x32_bf16(
                  false, a, false, bm, (short)0, acc, false, false);
      }
      const int cb = mt * 16 + half * 8;
#pragma unroll
      for (int r = 0; r < 8; ++r) {
        const size_t idx = ((size_t)(b * CC + cb + r) * LL + t) * HWSZ + pcol;
        const float xv = __builtin_nontemporal_load(&x[idx]);
        __builtin_nontemporal_store(xv + acc[r], &out[idx]);
      }
    }

    // ---- stage tile t+1 into the alternate buffer ----
    if (t + 1 < LL) {
      const float dtn = listT[b * LL + t + 1];
#pragma unroll
      for (int j = 0; j < 4; ++j) sA[nxt][w][lane * 4 + j] = __expf(-drl[j] * dtn);
#pragma unroll
      for (int j = 0; j < 16; ++j) {
        const int c = j * 8 + c_off;
        sXT[nxt][w][px4 + 0][c] = f2bf(rx[j].x);
        sXT[nxt][w][px4 + 1][c] = f2bf(rx[j].y);
        sXT[nxt][w][px4 + 2][c] = f2bf(rx[j].z);
        sXT[nxt][w][px4 + 3][c] = f2bf(rx[j].w);
      }
    }
    __syncthreads();  // staged tile visible for next step
  }

  // ---- final hidden state (streaming stores) ----
#pragma unroll
  for (int mt = 0; mt < 8; ++mt) {
    const int cb = mt * 16 + half * 8;
#pragma unroll
    for (int r = 0; r < 8; ++r)
      __builtin_nontemporal_store(h[mt][r],
          &hout[(size_t)(b * CC + cb + r) * HWSZ + pcol]);
  }
}

extern "C" void kernel_launch(void* const* d_in, const int* in_sizes, int n_in,
                              void* d_out, int out_size, void* d_ws, size_t ws_size,
                              hipStream_t stream) {
  const float* x     = (const float*)d_in[0];
  const float* lh    = (const float*)d_in[1];
  const float* listT = (const float*)d_in[2];
  const float* lam   = (const float*)d_in[3];
  const float* U     = (const float*)d_in[4];
  const float* V     = (const float*)d_in[5];
  const float* Wt    = (const float*)d_in[6];

  unsigned short* wsM = (unsigned short*)d_ws;        // 32*32*16 bf16 = 32KB
  unsigned short* wsW = wsM + 32 * 32 * 16;           // 32KB
  float*          dr  = (float*)(wsW + 32 * 32 * 16); // 512B

  float* out  = (float*)d_out;
  float* hout = out + (size_t)BB * CC * LL * HWSZ;

  prl_prep<<<1, 256, 0, stream>>>(lam, U, V, Wt, wsM, wsW, dr);
  prl_main<<<BB * (HWSZ / NT), 128, 0, stream>>>(x, lh, listT, wsM, wsW, dr, out, hout);
}
